// LocalAttention__32641751449858
// MI455X (gfx1250) — matrix operand order repaired and checked by reading, not compiled, not run on hardware
//
#include <hip/hip_runtime.h>

#define WIN   128
#define KLEN  256        // backward window + current window
#define EDIM  64
#define TLEN  4096
#define NWIN  32         // 4096 / 128
#define NBAT  32         // 2 * 16 merged
#define NWAVES 8

typedef __bf16 v16bf __attribute__((ext_vector_type(16)));
typedef __bf16 v8bf  __attribute__((ext_vector_type(8)));
typedef float  v8f   __attribute__((ext_vector_type(8)));

__device__ __forceinline__ void cvt8(v16bf& a, int o, float4 lo, float4 hi) {
    a[o + 0] = (__bf16)lo.x; a[o + 1] = (__bf16)lo.y;
    a[o + 2] = (__bf16)lo.z; a[o + 3] = (__bf16)lo.w;
    a[o + 4] = (__bf16)hi.x; a[o + 5] = (__bf16)hi.y;
    a[o + 6] = (__bf16)hi.z; a[o + 7] = (__bf16)hi.w;
}

__device__ __forceinline__ float hmax16(float v) {
    v = fmaxf(v, __shfl_xor(v, 1));
    v = fmaxf(v, __shfl_xor(v, 2));
    v = fmaxf(v, __shfl_xor(v, 4));
    v = fmaxf(v, __shfl_xor(v, 8));
    return v;
}
__device__ __forceinline__ float hsum16(float v) {
    v += __shfl_xor(v, 1);
    v += __shfl_xor(v, 2);
    v += __shfl_xor(v, 4);
    v += __shfl_xor(v, 8);
    return v;
}

__global__ __launch_bounds__(256)
void local_attn_wmma(const float* __restrict__ q,  const float* __restrict__ k,
                     const float* __restrict__ qr, const float* __restrict__ kr,
                     const float* __restrict__ v,  float* __restrict__ out)
{
    __shared__ __bf16 sK [KLEN][EDIM];          // K  row-major   (32 KB)
    __shared__ __bf16 sKr[KLEN][EDIM];          // K_rot          (32 KB)
    __shared__ __bf16 sVt[EDIM][KLEN];          // V transposed   (32 KB)
    __shared__ __bf16 sP [NWAVES][16][KLEN];    // probs per wave (64 KB)

    const int bw   = blockIdx.x;
    const int b    = bw / NWIN;
    const int w    = bw % NWIN;
    const int tid  = threadIdx.x;
    const int wave = tid >> 5;
    const int lane = tid & 31;
    const int half = lane >> 4;     // which 16-lane group
    const int lr   = lane & 15;

    // ---------------- stage K / K_rot / V(T) into LDS as bf16 ----------------
    {
        const int  key  = tid;                        // one key row per thread
        const int  t0   = (w - 1) * WIN + key;        // absolute time of key
        const bool pad  = (w == 0) && (key < WIN);    // zero-padded backward window
        const size_t gb = ((size_t)b * TLEN + t0) * EDIM;
        #pragma unroll
        for (int c = 0; c < EDIM; c += 4) {
            float4 fk, fr, fv;
            if (!pad) {
                fk = *(const float4*)(k  + gb + c);
                fr = *(const float4*)(kr + gb + c);
                fv = *(const float4*)(v  + gb + c);
            } else {
                fk = make_float4(0.f, 0.f, 0.f, 0.f); fr = fk; fv = fk;
            }
            sK [key][c+0] = (__bf16)fk.x; sK [key][c+1] = (__bf16)fk.y;
            sK [key][c+2] = (__bf16)fk.z; sK [key][c+3] = (__bf16)fk.w;
            sKr[key][c+0] = (__bf16)fr.x; sKr[key][c+1] = (__bf16)fr.y;
            sKr[key][c+2] = (__bf16)fr.z; sKr[key][c+3] = (__bf16)fr.w;
            sVt[c+0][key] = (__bf16)fv.x; sVt[c+1][key] = (__bf16)fv.y;
            sVt[c+2][key] = (__bf16)fv.z; sVt[c+3][key] = (__bf16)fv.w;
        }
    }
    __syncthreads();

    // ---------------- Q / Q_rot into WMMA A-layout registers ----------------
    // A 16x32 bf16, contiguous k order used here: lane<16 holds K = s*32 + {0..15},
    //                      lane>=16 holds K = s*32 + {16..31}, row = lane%16
    const int    qrow = w * WIN + wave * 16 + lr;     // absolute query time (A row)
    const float* qp   = q  + ((size_t)b * TLEN + qrow) * EDIM;
    const float* qrp  = qr + ((size_t)b * TLEN + qrow) * EDIM;

    v16bf aq[2], aqr[2];
    #pragma unroll
    for (int s = 0; s < 2; ++s) {
        const int b0 = s * 32 + half * 16;
        const int b1 = b0 + 8;
        cvt8(aq[s], 0, *(const float4*)(qp  + b0), *(const float4*)(qp  + b0 + 4));
        cvt8(aq[s], 8, *(const float4*)(qp  + b1), *(const float4*)(qp  + b1 + 4));
        cvt8(aqr[s], 0, *(const float4*)(qrp + b0), *(const float4*)(qrp + b0 + 4));
        cvt8(aqr[s], 8, *(const float4*)(qrp + b1), *(const float4*)(qrp + b1 + 4));
    }

    // ---------------- scores: 16 key-tiles of 16, dots + dots_rot -----------
    float sc[16][8];
    const int qkrel = WIN + wave * 16 + 8 * half;   // key-relative query pos (+r)
    #pragma unroll
    for (int j = 0; j < 16; ++j) {
        const int key = j * 16 + lr;                // B column = key (C column too)
        // B 32x16 bf16, same contiguous k order as A: lane<16 K=0..15, lane>=16 K=16..31, col = lane%16
        const v16bf bk0 = *(const v16bf*)&sK [key][      half * 16];
        const v16bf bk1 = *(const v16bf*)&sK [key][32 +  half * 16];
        const v16bf br0 = *(const v16bf*)&sKr[key][      half * 16];
        const v16bf br1 = *(const v16bf*)&sKr[key][32 +  half * 16];
        v8f c = {};
        c = __builtin_amdgcn_wmma_f32_16x16x32_bf16(false, aq[0],  false, bk0, (short)0, c, false, false);
        c = __builtin_amdgcn_wmma_f32_16x16x32_bf16(false, aq[1],  false, bk1, (short)0, c, false, false);
        c = __builtin_amdgcn_wmma_f32_16x16x32_bf16(false, aqr[0], false, br0, (short)0, c, false, false);
        c = __builtin_amdgcn_wmma_f32_16x16x32_bf16(false, aqr[1], false, br1, (short)0, c, false, false);
        #pragma unroll
        for (int r = 0; r < 8; ++r) {               // C row m = r + 8*half
            const bool masked = (key > qkrel + r) || ((w == 0) && (key < WIN));
            sc[j][r] = masked ? -1.0e30f : c[r] * 0.125f;
        }
    }

    // ---------------- softmax over 256 keys per row --------------------------
    #pragma unroll
    for (int r = 0; r < 8; ++r) {
        float mx = -1.0e30f;
        #pragma unroll
        for (int j = 0; j < 16; ++j) mx = fmaxf(mx, sc[j][r]);
        mx = hmax16(mx);
        float sum = 0.f;
        #pragma unroll
        for (int j = 0; j < 16; ++j) { float e = __expf(sc[j][r] - mx); sc[j][r] = e; sum += e; }
        sum = hsum16(sum);
        const float inv = 1.0f / sum;
        const int m = r + 8 * half;
        #pragma unroll
        for (int j = 0; j < 16; ++j)
            sP[wave][m][j * 16 + lr] = (__bf16)(sc[j][r] * inv);
    }
    __syncthreads();   // order LDS writes vs. cross-lane A-layout reloads

    // ---------------- out = P (16x256) @ V (256x64) ---------------------------
    #pragma unroll
    for (int nt = 0; nt < 4; ++nt) {
        v8f acc = {};
        #pragma unroll
        for (int ks = 0; ks < 8; ++ks) {
            // A-layout reload of P: two 8-element (16 B) LDS chunks per lane
            const v8bf p0 = *(const v8bf*)&sP[wave][lr][ks * 32 +     half * 16];
            const v8bf p1 = *(const v8bf*)&sP[wave][lr][ks * 32 + 8 + half * 16];
            v16bf pa;
            #pragma unroll
            for (int i = 0; i < 8; ++i) { pa[i] = p0[i]; pa[8 + i] = p1[i]; }
            const v16bf vb = *(const v16bf*)&sVt[nt * 16 + lr][ks * 32 + half * 16];
            acc = __builtin_amdgcn_wmma_f32_16x16x32_bf16(false, pa, false, vb, (short)0, acc, false, false);
        }
        #pragma unroll
        for (int r = 0; r < 8; ++r) {
            const int t = w * WIN + wave * 16 + r + 8 * half;
            out[((size_t)b * TLEN + t) * EDIM + nt * 16 + lr] = acc[r];
        }
    }
}

extern "C" void kernel_launch(void* const* d_in, const int* in_sizes, int n_in,
                              void* d_out, int out_size, void* d_ws, size_t ws_size,
                              hipStream_t stream) {
    (void)in_sizes; (void)n_in; (void)out_size; (void)d_ws; (void)ws_size;
    const float* q  = (const float*)d_in[0];
    const float* k  = (const float*)d_in[1];
    const float* qr = (const float*)d_in[2];
    const float* kr = (const float*)d_in[3];
    const float* v  = (const float*)d_in[4];
    float* out = (float*)d_out;
    dim3 grid(NBAT * NWIN);   // 32 batches * 32 windows = 1024 tiles
    dim3 block(256);          // 8 waves; one 128-query-row tile per workgroup
    local_attn_wmma<<<grid, block, 0, stream>>>(q, k, qr, kr, v, out);
}
